// RNNPosEncoder_13151189860703
// MI455X (gfx1250) — matrix operand order repaired and checked by reading, not compiled, not run on hardware
//
#include <hip/hip_runtime.h>

typedef __attribute__((ext_vector_type(16))) _Float16 v16h;
typedef __attribute__((ext_vector_type(8)))  float    v8f;

#define DEPTH   32
#define WIDTH   64
#define HID     50
#define GATES   150          // 3*HID
#define EMB     768
#define NMATH   50000
#define NTILES  (NMATH/16)   // 3125 exactly
#define WAVES   4
#define GS      165          // padded row stride for gx/gh scratch (spreads banks)
#define HS      68           // padded row stride for h scratch (16B aligned rows)

// d_ws layout (in _Float16 units): WMMA B-fragments, [tile][kchunk][lane][16 halves]
#define WIH_OFF   0
#define WIH_HALF  (10*2*32*16)   // 10240  (W_ih: N=160 -> 10 tiles, K=64 -> 2 chunks)
#define WHH_OFF   (WIH_OFF + WIH_HALF)
#define WHH_HALF  (10*2*32*16)   // 10240  (W_hh: K padded 50->64)
#define WOUT_OFF  (WHH_OFF + WHH_HALF)          // 20480
#define WOUT_HALF (48*2*32*16)   // 49152  (W_out: N=768 -> 48 tiles)
#define WS_HALVES (WOUT_OFF + WOUT_HALF)        // 69632 halves = 136 KB

#define WMMA_F16(A,B,C) \
  __builtin_amdgcn_wmma_f32_16x16x32_f16(false,(A),false,(B),(short)0,(C),false,false)

// ---------------------------------------------------------------------------
// Prep: convert fp32 weights into f16 WMMA B-fragment layout in workspace.
// B fragment layout (16-bit, 32x16 K x N chunk), per ISA 7.12.2:
//   element (k,n): lane = (n%16) + 16*((k%16)/8), halfIdx e = (k%8) + 8*((k%32)/16)
// ---------------------------------------------------------------------------
__global__ void prep_weights(const float* __restrict__ W_ih,
                             const float* __restrict__ W_hh,
                             const float* __restrict__ W_out,
                             _Float16* __restrict__ ws) {
  int id = blockIdx.x * 256 + threadIdx.x;
  if (id >= WS_HALVES) return;
  int arr, idx;
  if (id < WHH_OFF)       { arr = 0; idx = id; }
  else if (id < WOUT_OFF) { arr = 1; idx = id - WHH_OFF; }
  else                    { arr = 2; idx = id - WOUT_OFF; }
  int e    = idx & 15;
  int lane = (idx >> 4) & 31;
  int c    = (idx >> 9) & 1;      // K chunk of 32
  int j    = idx >> 10;           // N tile of 16
  int n    = j * 16 + (lane & 15);
  int k    = 32 * c + 8 * (lane >> 4) + (e & 7) + 16 * (e >> 3);
  float v = 0.f;
  if (arr == 0)      { if (n < GATES)            v = W_ih[n * WIDTH + k]; }
  else if (arr == 1) { if (n < GATES && k < HID) v = W_hh[n * HID + k]; }
  else               { if (k < HID)              v = W_out[n * HID + k]; }
  ws[id] = (_Float16)v;
}

// A-fragment loader (16-bit A, 16x32 chunk). Lane holds row m=lane%16 and
// K groups {kbase..kbase+7} and {kbase+16..kbase+23} where kbase=32c+8*(lane/16).
// All offsets 16B aligned -> b128 loads (global or ds after addrspace inference).
__device__ __forceinline__ v16h load_a16(const float* __restrict__ row, int kbase) {
  const float4 p0 = *(const float4*)(row + kbase);
  const float4 p1 = *(const float4*)(row + kbase + 4);
  const float4 q0 = *(const float4*)(row + kbase + 16);
  const float4 q1 = *(const float4*)(row + kbase + 20);
  v16h a;
  a[0]=(_Float16)p0.x;  a[1]=(_Float16)p0.y;  a[2]=(_Float16)p0.z;  a[3]=(_Float16)p0.w;
  a[4]=(_Float16)p1.x;  a[5]=(_Float16)p1.y;  a[6]=(_Float16)p1.z;  a[7]=(_Float16)p1.w;
  a[8]=(_Float16)q0.x;  a[9]=(_Float16)q0.y;  a[10]=(_Float16)q0.z; a[11]=(_Float16)q0.w;
  a[12]=(_Float16)q1.x; a[13]=(_Float16)q1.y; a[14]=(_Float16)q1.z; a[15]=(_Float16)q1.w;
  return a;
}

__device__ __forceinline__ float sigmoidf_fast(float x) {
  return 1.f / (1.f + __expf(-x));
}

// Wave-internal LDS ordering: DS ops from one wave complete in order; the wait
// plus memory clobber makes cross-lane LDS results visible & blocks reordering.
#define WAVE_LDS_SYNC() asm volatile("s_wait_dscnt 0x0" ::: "memory")

// ---------------------------------------------------------------------------
// Fused: gather -> gx GEMM -> 32-step GRU recurrence -> output projection.
// One wave per 16-sample tile, 4 waves/block. 144KB LDS -> ~2 blocks/WGP, so
// latency hiding comes from ILP: next x row prefetched, and B-fragments are
// software-pipelined one N-tile ahead of their WMMAs.
// ---------------------------------------------------------------------------
__launch_bounds__(128, 1)
__global__ void rnn_pos_encoder(const float* __restrict__ pos,
                                const float* __restrict__ b_ih,
                                const float* __restrict__ b_hh,
                                const float* __restrict__ b_out,
                                const int*   __restrict__ math_idxs,
                                const int*   __restrict__ pos_levels,
                                const _Float16* __restrict__ ws,
                                float* __restrict__ out) {
  __shared__ __align__(64) _Float16 s_wfrag[WIH_HALF + WHH_HALF]; // 40 KB
  __shared__ float s_bih[GATES];
  __shared__ float s_bhh[GATES];
  __shared__ float s_bout[EMB];
  __shared__ __align__(64) float s_gx[WAVES][16][GS];
  __shared__ __align__(64) float s_gh[WAVES][16][GS];
  __shared__ __align__(64) float s_h [WAVES][16][HS];

  const int tid = threadIdx.x;

  // Cooperative fills (all 128 threads, before any wave may exit).
  {
    const unsigned int* src = (const unsigned int*)ws;
    unsigned int* dst = (unsigned int*)s_wfrag;
    for (int i = tid; i < (WIH_HALF + WHH_HALF) / 2; i += 128) dst[i] = src[i];
    for (int i = tid; i < GATES; i += 128) { s_bih[i] = b_ih[i]; s_bhh[i] = b_hh[i]; }
    for (int i = tid; i < EMB;   i += 128) s_bout[i] = b_out[i];
    float* hz = &s_h[0][0][0];
    for (int i = tid; i < WAVES * 16 * HS; i += 128) hz[i] = 0.f; // h0=0; K-pad cols stay 0
  }
  __syncthreads();

  const int wave = tid >> 5;
  const int lane = tid & 31;
  const int tile = blockIdx.x * WAVES + wave;
  if (tile >= NTILES) return;   // uniform per wave; EXEC stays all-ones for WMMA

  const int m  = lane & 15;     // sample row within tile
  const int hf = lane >> 4;     // lane-half selector in fragment layouts
  const int samp = tile * 16 + m;
  const int gidx = math_idxs[samp];
  const int len  = pos_levels[gidx] + 1;
  const float* xbase = pos + (size_t)gidx * (DEPTH * WIDTH);

  const v16h* whiF  = (const v16h*)s_wfrag;               // fragment f at whiF[f*32+lane]
  const v16h* whhF  = (const v16h*)(s_wfrag + WIH_HALF);
  const v16h* woutF = (const v16h*)(ws + WOUT_OFF);       // L2-resident, 96 KB

  float* gxw = &s_gx[wave][0][0];
  float* ghw = &s_gh[wave][0][0];
  float* hw  = &s_h [wave][0][0];
  const float* hrowA = hw + m * HS;     // A-frag source row for h
  const int k0 = 8 * hf;                // chunk-0 K base for this lane

  // Pull first gathered row toward the caches (global_prefetch_b8).
  __builtin_prefetch(xbase + k0, 0, 0);
  __builtin_prefetch(xbase + 32 + k0, 0, 0);

#pragma unroll 1
  for (int t = 0; t < DEPTH; ++t) {
    // A-fragments of x_t (16x64 -> two 16x32 chunks), gathered, 16B-aligned loads
    const float* xrow = xbase + t * WIDTH;
    const v16h xa0 = load_a16(xrow, k0);
    const v16h xa1 = load_a16(xrow, 32 + k0);

    // Prefetch next step's gathered row: hides HBM latency behind this step's
    // 40 WMMAs + gate math (low occupancy -> prefetch matters).
    if (t + 1 < DEPTH) {
      __builtin_prefetch(xrow + WIDTH + k0, 0, 0);
      __builtin_prefetch(xrow + WIDTH + 32 + k0, 0, 0);
    }

    // A-fragments of h (cols 50..63 are zero -> matches zero-padded K of W_hh)
    WAVE_LDS_SYNC();                        // see prior gate-phase writes
    const v16h ha0 = load_a16(hrowA, k0);
    const v16h ha1 = load_a16(hrowA, 32 + k0);

    // Software-pipelined B-fragments: tile j+1's 4 fragments are issued before
    // tile j's WMMAs, so (with in-order DS completion) the pre-WMMA waits can
    // tolerate outstanding loads instead of draining DScnt to 0 every WMMA.
    v16h b0 = whiF[0 * 32 + lane];
    v16h b1 = whiF[1 * 32 + lane];
    v16h c0 = whhF[0 * 32 + lane];
    v16h c1 = whhF[1 * 32 + lane];
#pragma unroll
    for (int j = 0; j < 10; ++j) {          // 10 N-tiles cover 150 gate columns
      v16h nb0, nb1, nc0, nc1;
      if (j < 9) {                          // compile-time after full unroll
        nb0 = whiF[((j + 1) * 2 + 0) * 32 + lane];
        nb1 = whiF[((j + 1) * 2 + 1) * 32 + lane];
        nc0 = whhF[((j + 1) * 2 + 0) * 32 + lane];
        nc1 = whhF[((j + 1) * 2 + 1) * 32 + lane];
      }
      v8f ax = {}; v8f ah = {};
      ax = WMMA_F16(xa0, b0, ax);
      ah = WMMA_F16(ha0, c0, ah);
      ax = WMMA_F16(xa1, b1, ax);
      ah = WMMA_F16(ha1, c1, ah);
      const int nn = j * 16 + m;            // C/D layout: lane holds n = lane%16
#pragma unroll
      for (int r = 0; r < 8; ++r) {         // C/D layout: VGPR r -> row r + 8*(lane/16)
        gxw[(r + 8 * hf) * GS + nn] = ax[r];
        ghw[(r + 8 * hf) * GS + nn] = ah[r];
      }
      if (j < 9) { b0 = nb0; b1 = nb1; c0 = nc0; c1 = nc1; }
    }
    WAVE_LDS_SYNC();

    // Gate phase: 16 samples x 50 units over 32 lanes (25 units per lane).
    const float* gx_r = gxw + m * GS;
    const float* gh_r = ghw + m * GS;
    float* hrow = hw + m * HS;
    for (int ii = 0; ii < 25; ++ii) {
      const int i = hf * 25 + ii;
      const float r = sigmoidf_fast(gx_r[i]       + gh_r[i]       + s_bih[i]       + s_bhh[i]);
      const float z = sigmoidf_fast(gx_r[50 + i]  + gh_r[50 + i]  + s_bih[50 + i]  + s_bhh[50 + i]);
      const float n = tanhf(gx_r[100 + i] + s_bih[100 + i] + r * (gh_r[100 + i] + s_bhh[100 + i]));
      const float hold = hrow[i];
      const float hnew = (1.f - z) * n + z * hold;
      hrow[i] = (t < len) ? hnew : hold;    // freeze after sample's length
    }
  }

  // Output projection: out = h @ W_out^T + b_out  (K padded 50->64, zeros).
  // Unroll x2 -> independent WMMA chains; global fragment loads pipeline on
  // LOADcnt across iterations.
  WAVE_LDS_SYNC();
  const v16h ha0 = load_a16(hrowA, k0);
  const v16h ha1 = load_a16(hrowA, 32 + k0);
  const int mbase = tile * 16 + 8 * hf;
#pragma unroll 2
  for (int j = 0; j < 48; ++j) {
    v8f acc = {};
    acc = WMMA_F16(ha0, woutF[(j * 2 + 0) * 32 + lane], acc);
    acc = WMMA_F16(ha1, woutF[(j * 2 + 1) * 32 + lane], acc);
    const int n = j * 16 + m;
    const float bo = s_bout[n];
#pragma unroll
    for (int r = 0; r < 8; ++r)
      out[(size_t)(mbase + r) * EMB + n] = acc[r] + bo;
  }
}

// ---------------------------------------------------------------------------
extern "C" void kernel_launch(void* const* d_in, const int* in_sizes, int n_in,
                              void* d_out, int out_size, void* d_ws, size_t ws_size,
                              hipStream_t stream) {
  const float* pos   = (const float*)d_in[0];
  const float* W_ih  = (const float*)d_in[1];
  const float* W_hh  = (const float*)d_in[2];
  const float* b_ih  = (const float*)d_in[3];
  const float* b_hh  = (const float*)d_in[4];
  const float* W_out = (const float*)d_in[5];
  const float* b_out = (const float*)d_in[6];
  const int* midx    = (const int*)d_in[7];
  const int* plev    = (const int*)d_in[8];
  _Float16* ws = (_Float16*)d_ws;     // needs 136 KB of workspace
  float* out = (float*)d_out;

  prep_weights<<<(WS_HALVES + 255) / 256, 256, 0, stream>>>(W_ih, W_hh, W_out, ws);

  const int nblocks = (NTILES + WAVES - 1) / WAVES;   // 782
  rnn_pos_encoder<<<nblocks, 128, 0, stream>>>(pos, b_ih, b_hh, b_out,
                                               midx, plev, ws, out);
}
